// SoftTokenMap_56882546868727
// MI455X (gfx1250) — matrix operand order repaired and checked by reading, not compiled, not run on hardware
//
#include <hip/hip_runtime.h>
#include <hip/hip_bf16.h>

typedef __bf16 bf16_t;
typedef __attribute__((ext_vector_type(16))) __bf16 v16bf;
typedef __attribute__((ext_vector_type(8)))  __bf16 v8bf;
typedef __attribute__((ext_vector_type(8)))  float  v8f;

#define ASR_VOCAB 1024
#define LLM_VOCAB 32000
#define EMBED_DIM 4096
#define NTOK      4096   // 8 * 512

// ---------------------------------------------------------------------------
// Kernel 1: row softmax of mapping_logits -> bf16 weight table (1024 x 32000)
// 1 block per row, 256 threads. 32000 = 256*125 exactly.
// ---------------------------------------------------------------------------
__global__ __launch_bounds__(256)
void softmax_bf16_kernel(const float* __restrict__ logits,
                         bf16_t* __restrict__ w16) {
  __shared__ float red[256];
  const int row = blockIdx.x;
  const int tid = threadIdx.x;
  const float* src = logits + (size_t)row * LLM_VOCAB;

  // pass 1: row max
  float m = -3.4e38f;
  for (int i = tid; i < LLM_VOCAB; i += 256) m = fmaxf(m, src[i]);
  red[tid] = m; __syncthreads();
  for (int s = 128; s > 0; s >>= 1) {
    if (tid < s) red[tid] = fmaxf(red[tid], red[tid + s]);
    __syncthreads();
  }
  m = red[0];
  __syncthreads();

  // pass 2: row sum of exp
  float sum = 0.f;
  for (int i = tid; i < LLM_VOCAB; i += 256) sum += __expf(src[i] - m);
  red[tid] = sum; __syncthreads();
  for (int s = 128; s > 0; s >>= 1) {
    if (tid < s) red[tid] += red[tid + s];
    __syncthreads();
  }
  const float inv = 1.f / red[0];

  // pass 3: normalized weights -> bf16 (bf16 keeps range; mean weight ~3e-5
  // would be subnormal in fp16)
  bf16_t* dst = w16 + (size_t)row * LLM_VOCAB;
  for (int i = tid; i < LLM_VOCAB; i += 256)
    dst[i] = (bf16_t)(__expf(src[i] - m) * inv);
}

// ---------------------------------------------------------------------------
// Kernel 2: T[1024][4096] = W16[1024][32000] x embed[32000][4096]
// bf16 WMMA (v_wmma_f32_16x16x32_bf16), f32 accumulate.
// Block tile 128x128, BK=32. 256 threads = 8 waves (2x4); each wave: 64x32.
// ---------------------------------------------------------------------------
#define BM 128
#define BN 128
#define BK 32
#define LDSS (BK + 16)   // bf16 elements: 48 -> 96-byte row stride (16B aligned)

__global__ __launch_bounds__(256)
void wmma_gemm_kernel(const bf16_t* __restrict__ w16,
                      const float* __restrict__ embed,
                      float* __restrict__ table) {
  __shared__ bf16_t As[BM * LDSS];   // [row][k]   12 KB
  __shared__ bf16_t Bs[BN * LDSS];   // [col][k]   12 KB (B transposed)

  const int tid  = threadIdx.x;
  const int lane = tid & 31;
  const int wave = tid >> 5;     // 0..7
  const int wm   = wave & 1;     // wave row (2)
  const int wn   = wave >> 1;    // wave col (4)
  const int lr   = lane & 15;    // lane row/col within 16
  const int kh   = lane >> 4;    // K-half select (ISA A/B lane layout)

  const int bm = blockIdx.x * BM;   // x fastest: M-blocks sharing the embed
  const int bn = blockIdx.y * BN;   // slab run adjacently -> L2 reuse

  const v8f vzero = {0.f,0.f,0.f,0.f,0.f,0.f,0.f,0.f};
  v8f acc[4][2];
  #pragma unroll
  for (int mf = 0; mf < 4; ++mf)
    #pragma unroll
    for (int nf = 0; nf < 2; ++nf) acc[mf][nf] = vzero;

  // global->LDS mapping
  const int a_r = tid >> 1;      // A: row 0..127
  const int a_h = tid & 1;       //    16-element K half (adjacent threads = 64B)
  const int b_n = tid & 127;     // B: output column 0..127
  const int b_h = tid >> 7;      //    K half 0..1

  for (int k0 = 0; k0 < LLM_VOCAB; k0 += BK) {
    // ---- A tile: 128x32 bf16, already bf16 in the L2-resident table ----
    {
      const uint4* ag = (const uint4*)(w16 + (size_t)(bm + a_r) * LLM_VOCAB
                                           + k0 + a_h * 16);
      uint4 v0 = ag[0];
      uint4 v1 = ag[1];
      *(uint4*)&As[a_r * LDSS + a_h * 16]     = v0;
      *(uint4*)&As[a_r * LDSS + a_h * 16 + 8] = v1;
    }
    // ---- B tile: 32x128 fp32 -> bf16, transposed into [col][k] ----
    {
      const float* bg = embed + (size_t)(k0 + b_h * 16) * EMBED_DIM + bn + b_n;
      __builtin_prefetch(bg + (size_t)BK * EMBED_DIM, 0, 0);  // next slab
      bf16_t tmp[16];
      #pragma unroll
      for (int j = 0; j < 16; ++j)
        tmp[j] = (bf16_t)bg[(size_t)j * EMBED_DIM];
      *(uint4*)&Bs[b_n * LDSS + b_h * 16]     = *(uint4*)&tmp[0];
      *(uint4*)&Bs[b_n * LDSS + b_h * 16 + 8] = *(uint4*)&tmp[8];
    }
    __syncthreads();

    // ---- fragments per ISA VGPR layouts ----
    // A 16x32 bf16: lane l -> row lr, VGPR0-3: K = kh*8..kh*8+7,
    //                              VGPR4-7: K = 16+kh*8..16+kh*8+7
    v16bf afrag[4];
    #pragma unroll
    for (int mf = 0; mf < 4; ++mf) {
      const int m = wm * 64 + mf * 16 + lr;
      v8bf lo = *(const v8bf*)&As[m * LDSS + kh * 8];
      v8bf hi = *(const v8bf*)&As[m * LDSS + 16 + kh * 8];
      afrag[mf] = __builtin_shufflevector(lo, hi,
          0,1,2,3,4,5,6,7,8,9,10,11,12,13,14,15);
    }
    // B 32x16 bf16: lane l -> col lr, VGPR0-3: K = kh*8..kh*8+7, VGPR4-7: K = 16+kh*8..16+kh*8+7
    v16bf bfrag[2];
    #pragma unroll
    for (int nf = 0; nf < 2; ++nf) {
      const int n = wn * 32 + nf * 16 + lr;
      v8bf lo = *(const v8bf*)&Bs[n * LDSS + kh * 8];
      v8bf hi = *(const v8bf*)&Bs[n * LDSS + 16 + kh * 8];
      bfrag[nf] = __builtin_shufflevector(lo, hi,
          0,1,2,3,4,5,6,7,8,9,10,11,12,13,14,15);
    }

    #pragma unroll
    for (int mf = 0; mf < 4; ++mf)
      #pragma unroll
      for (int nf = 0; nf < 2; ++nf)
        acc[mf][nf] = __builtin_amdgcn_wmma_f32_16x16x32_bf16(
            false, afrag[mf], false, bfrag[nf],
            (short)0, acc[mf][nf], false, false);
    __syncthreads();
  }

  // ---- epilogue: C/D layout: VGPR j, lanes 0-15 -> M=j, lanes 16-31 -> M=j+8
  #pragma unroll
  for (int mf = 0; mf < 4; ++mf) {
    #pragma unroll
    for (int nf = 0; nf < 2; ++nf) {
      const int col   = bn + wn * 32 + nf * 16 + lr;
      const int rbase = bm + wm * 64 + mf * 16 + kh * 8;
      #pragma unroll
      for (int j = 0; j < 8; ++j)
        table[(size_t)(rbase + j) * EMBED_DIM + col] = acc[mf][nf][j];
    }
  }
}

// ---------------------------------------------------------------------------
// Kernel 3: out[tok] = table[token_ids[tok]]  (4096 rows x 4096 f32)
// ---------------------------------------------------------------------------
__global__ __launch_bounds__(256)
void gather_kernel(const int* __restrict__ ids,
                   const float* __restrict__ table,
                   float* __restrict__ out) {
  const int tok = blockIdx.x;
  const int row = ids[tok];
  const float4* src = (const float4*)(table + (size_t)row * EMBED_DIM);
  float4*       dst = (float4*)(out + (size_t)tok * EMBED_DIM);
  for (int i = threadIdx.x; i < EMBED_DIM / 4; i += 256)
    dst[i] = src[i];
}

// ---------------------------------------------------------------------------
extern "C" void kernel_launch(void* const* d_in, const int* in_sizes, int n_in,
                              void* d_out, int out_size, void* d_ws, size_t ws_size,
                              hipStream_t stream) {
  (void)in_sizes; (void)n_in; (void)out_size; (void)ws_size;
  const int*   token_ids = (const int*)  d_in[0];   // (8,512) int32
  const float* logits    = (const float*)d_in[1];   // (1024,32000) f32
  const float* embed     = (const float*)d_in[2];   // (32000,4096) f32
  float*       out       = (float*)d_out;           // (8,512,4096) f32

  // workspace: [ bf16 weights 65.5MB | f32 table 16.8MB ]
  bf16_t* w16   = (bf16_t*)d_ws;
  float*  table = (float*)((char*)d_ws +
                           (size_t)ASR_VOCAB * LLM_VOCAB * sizeof(bf16_t));

  softmax_bf16_kernel<<<ASR_VOCAB, 256, 0, stream>>>(logits, w16);

  dim3 grid(ASR_VOCAB / BM, EMBED_DIM / BN);   // (8, 32)
  wmma_gemm_kernel<<<grid, 256, 0, stream>>>(w16, embed, table);

  gather_kernel<<<NTOK, 256, 0, stream>>>(token_ids, table, out);
}
